// ExtenderAttention_9062380995186
// MI455X (gfx1250) — matrix operand order repaired and checked by reading, not compiled, not run on hardware
//
#include <hip/hip_runtime.h>
#include <cstdint>
#include <cstddef>

// ---------------------------------------------------------------------------
// ExtenderAttention for MI455X (gfx1250, wave32, WMMA)
//   B=2 L=4096 D=1024 H=16 P=4 BS=128 HD=64 NB=32
// Pipeline:
//   1) k_gemm_proj   x3 : (B*L,D)@(D,D)+b -> (B,H,L,HD) f32, WMMA f16 tiles
//   2) k_lowrank         : sort keys q_key/k_key (B,H,P,L) with mask/BIG
//   3) k_bitonic_argsort : 256 blocks, 4096-element stable ascending argsort
//   4) k_block_attn      : per (b,h,p,nb): gather->WMMA scores->entmax1.5
//                          (register bisection, wave32 shuffles)->WMMA ctx
//                          ->atomic scatter-mean over P
//   5) k_finalize        : (B,H,L,HD)->(B,L,D), mask zeroing
// ---------------------------------------------------------------------------

#define B_  2
#define L_  4096
#define D_  1024
#define H_  16
#define P_  4
#define BS_ 128
#define HD_ 64
#define NB_ 32
#define BIGF 1000000000.0f

typedef __attribute__((ext_vector_type(16))) _Float16 v16h;
typedef __attribute__((ext_vector_type(8)))  float    v8f;

union FragU { v16h h; uint4 q[2]; };

// A fragment (16-bit A matrix 16x32, contiguous k order used by this code, not the ISA table order):
//  lane&15 = M, lane>>4 = half; elems e <- K = koff + 16*half + e
__device__ __forceinline__ v16h ld_frag_a(const _Float16* base, int strideH,
                                          int mrow, int koff) {
  const int lane = threadIdx.x & 31;
  const int m    = mrow + (lane & 15);
  const int half = lane >> 4;
  FragU f;
  f.q[0] = *(const uint4*)(base + m * strideH + koff + 16 * half);
  f.q[1] = *(const uint4*)(base + m * strideH + koff + 16 * half + 8);
  return f.h;
}

// B fragment (16-bit B matrix 32x16): lane&15 = N, elems e <- K = koff+16*half+e.
// Staged LDS layout is [N][K] so the 16 halfs are contiguous.
__device__ __forceinline__ v16h ld_frag_b(const _Float16* base, int strideH,
                                          int ncol, int koff) {
  const int lane = threadIdx.x & 31;
  const int n    = ncol + (lane & 15);
  const uint4* p = (const uint4*)(base + n * strideH + koff + 16 * (lane >> 4));
  FragU f;
  f.q[0] = p[0];
  f.q[1] = p[1];
  return f.h;
}

// ---------------------------------------------------------------------------
// 1) Projection GEMM: Y(BL,D) = X(BL,D) @ W(D,D) + bias, stored as (B,H,L,HD)
//    128x128 tile / workgroup, 8 waves (4 M x 2 N), each wave 32x64 = 2x4 WMMA
// ---------------------------------------------------------------------------
__global__ __launch_bounds__(256) void k_gemm_proj(const float* __restrict__ X,
                                                   const float* __restrict__ W,
                                                   const float* __restrict__ bias,
                                                   float* __restrict__ out) {
  __shared__ __align__(16) _Float16 As[128 * 40];  // [m][k], stride 40 halfs
  __shared__ __align__(16) _Float16 Bs[128 * 40];  // [n][k], stride 40 halfs

  const int tid   = threadIdx.x;
  const int tileM = blockIdx.x * 128;
  const int tileN = blockIdx.y * 128;
  const int w  = tid >> 5;
  const int wm = w >> 1;  // 0..3 -> M offset 32*wm
  const int wn = w & 1;   // 0..1 -> N offset 64*wn

  v8f zero = {};
  v8f acc[2][4];
  for (int a = 0; a < 2; a++)
    for (int t = 0; t < 4; t++) acc[a][t] = zero;

  for (int k0 = 0; k0 < D_; k0 += 32) {
    // stage A: 128x32, f32->f16, 2 threads per row
    {
      const int r  = tid >> 1;
      const int c0 = (tid & 1) * 16;
      const float* src = X + (size_t)(tileM + r) * D_ + k0 + c0;
      if (k0 + 32 < D_) __builtin_prefetch(src + 32, 0, 0);
      _Float16* dst = As + r * 40 + c0;
      for (int j = 0; j < 16; j += 4) {
        float4 f = *(const float4*)(src + j);
        dst[j + 0] = (_Float16)f.x; dst[j + 1] = (_Float16)f.y;
        dst[j + 2] = (_Float16)f.z; dst[j + 3] = (_Float16)f.w;
      }
    }
    // stage B transposed: Bs[n][k] = W[k0+k][tileN+n]
    {
      const int k  = tid >> 3;
      const int n0 = (tid & 7) * 16;
      const float* src = W + (size_t)(k0 + k) * D_ + tileN + n0;
      for (int j = 0; j < 16; j += 4) {
        float4 f = *(const float4*)(src + j);
        Bs[(n0 + j + 0) * 40 + k] = (_Float16)f.x;
        Bs[(n0 + j + 1) * 40 + k] = (_Float16)f.y;
        Bs[(n0 + j + 2) * 40 + k] = (_Float16)f.z;
        Bs[(n0 + j + 3) * 40 + k] = (_Float16)f.w;
      }
    }
    __syncthreads();

    v16h a0 = ld_frag_a(As, 40, 32 * wm, 0);
    v16h a1 = ld_frag_a(As, 40, 32 * wm + 16, 0);
    for (int t = 0; t < 4; t++) {
      v16h bf = ld_frag_b(Bs, 40, 64 * wn + 16 * t, 0);
      acc[0][t] = __builtin_amdgcn_wmma_f32_16x16x32_f16(
          false, a0, false, bf, (short)0, acc[0][t], false, false);
      acc[1][t] = __builtin_amdgcn_wmma_f32_16x16x32_f16(
          false, a1, false, bf, (short)0, acc[1][t], false, false);
    }
    __syncthreads();
  }

  // epilogue: bias + store directly in (B,H,L,HD) layout
  const int lane = tid & 31;
  const int half = lane >> 4;
  for (int a = 0; a < 2; a++) {
    for (int t = 0; t < 4; t++) {
      const int col = tileN + 64 * wn + 16 * t + (lane & 15);
      const float bv = bias[col];
      const int h  = col >> 6;   // / HD_
      const int hd = col & 63;
      for (int i = 0; i < 8; i++) {
        const int row = tileM + 32 * wm + 16 * a + i + 8 * half;
        const int bb  = row >> 12;       // / L_  (tiles never straddle batch)
        const int l   = row & (L_ - 1);
        out[(((size_t)bb * H_ + h) * L_ + l) * HD_ + hd] = acc[a][t][i] + bv;
      }
    }
  }
}

// ---------------------------------------------------------------------------
// 2) Low-rank sort keys: q_key/k_key[b,h,p,l] = mask ? proj+bias : BIG
// ---------------------------------------------------------------------------
__global__ __launch_bounds__(256) void k_lowrank(const float* __restrict__ qh,
                                                 const float* __restrict__ kh,
                                                 const float* __restrict__ pwq,
                                                 const float* __restrict__ pbq,
                                                 const float* __restrict__ pwk,
                                                 const float* __restrict__ pbk,
                                                 const uint8_t* __restrict__ mask,
                                                 float* __restrict__ qkey,
                                                 float* __restrict__ kkey) {
  const int t = blockIdx.x * 256 + threadIdx.x;
  if (t >= B_ * H_ * L_) return;
  const int b = t / (H_ * L_);
  const int h = (t / L_) % H_;
  const int l = t % L_;
  const size_t base = (((size_t)b * H_ + h) * L_ + l) * HD_;

  float aq[P_] = {0.f, 0.f, 0.f, 0.f};
  float ak[P_] = {0.f, 0.f, 0.f, 0.f};
  for (int hd = 0; hd < HD_; hd++) {
    const float xq = qh[base + hd];
    const float xk = kh[base + hd];
    const float* wq = pwq + (h * HD_ + hd) * P_;
    const float* wk = pwk + (h * HD_ + hd) * P_;
    for (int p = 0; p < P_; p++) {
      aq[p] += xq * wq[p];
      ak[p] += xk * wk[p];
    }
  }
  const bool mv = mask[(size_t)b * L_ + l] != 0;
  for (int p = 0; p < P_; p++) {
    const size_t o = (((size_t)b * H_ + h) * P_ + p) * L_ + l;
    qkey[o] = mv ? (aq[p] + pbq[h * P_ + p]) : BIGF;
    kkey[o] = mv ? (ak[p] + pbk[h * P_ + p]) : BIGF;
  }
}

// ---------------------------------------------------------------------------
// 3) Bitonic argsort, ascending, stable (index tie-break). 1 block = 1 array.
// ---------------------------------------------------------------------------
__global__ __launch_bounds__(1024) void k_bitonic_argsort(const float* __restrict__ keys,
                                                          int* __restrict__ idxout) {
  __shared__ float sk[L_];
  __shared__ int   si[L_];
  const float* kb = keys + (size_t)blockIdx.x * L_;
  for (int i = threadIdx.x; i < L_; i += 1024) { sk[i] = kb[i]; si[i] = i; }
  __syncthreads();

  for (int k = 2; k <= L_; k <<= 1) {
    for (int j = k >> 1; j > 0; j >>= 1) {
      for (int q = 0; q < L_ / 1024; q++) {
        const int i   = threadIdx.x + q * 1024;
        const int ixj = i ^ j;
        if (ixj > i) {
          const bool up = ((i & k) == 0);
          const float a = sk[i], c = sk[ixj];
          const int  ia = si[i], ic = si[ixj];
          const bool cmp = (a > c) || (a == c && ia > ic);  // i-elem after j-elem
          if (cmp == up) {
            sk[i] = c; sk[ixj] = a;
            si[i] = ic; si[ixj] = ia;
          }
        }
      }
      __syncthreads();
    }
  }
  int* ob = idxout + (size_t)blockIdx.x * L_;
  for (int i = threadIdx.x; i < L_; i += 1024) ob[i] = si[i];
}

// ---------------------------------------------------------------------------
// 4) Block attention: one workgroup (8 waves) per (b,h,p,nb).
//    LDS (54784 B): qs[128][72]f16 | ks[128][72]f16 (aliased later by
//    attn[128][136]f16) ; vsT[64][136]f16 ; kmask[128]f32
// ---------------------------------------------------------------------------
__global__ __launch_bounds__(256) void k_block_attn(const float* __restrict__ qh,
                                                    const float* __restrict__ kh,
                                                    const float* __restrict__ vh,
                                                    const int* __restrict__ idxq,
                                                    const int* __restrict__ idxk,
                                                    const uint8_t* __restrict__ mask,
                                                    float* __restrict__ ctx) {
  __shared__ __align__(16) char smem[54784];
  _Float16* qs   = (_Float16*)smem;             // [128][72]   (18432 B)
  _Float16* ksS  = (_Float16*)(smem + 18432);   // [128][72]   (18432 B)
  _Float16* attn = (_Float16*)smem;             // [128][136]  aliases qs/ks
  _Float16* vsT  = (_Float16*)(smem + 36864);   // [64][136]   (17408 B)
  float*    kmf  = (float*)(smem + 54272);      // [128]

  const int tid = threadIdx.x;
  const int nb  = blockIdx.x & (NB_ - 1);
  const int p   = (blockIdx.x >> 5) & (P_ - 1);
  const int h   = (blockIdx.x >> 7) & (H_ - 1);
  const int b   = blockIdx.x >> 11;
  const size_t sbase = ((((size_t)b * H_ + h) * P_ + p) * L_) + (size_t)nb * BS_;

  // gather sorted rows: q by idxq, k/v by idxk; vs stored transposed [hd][key]
  {
    const int r  = tid >> 1;
    const int c0 = (tid & 1) * 32;
    const int lq = idxq[sbase + r];
    const int lk = idxk[sbase + r];
    const size_t hb = (((size_t)b * H_ + h) * L_);
    const float* qrow = qh + (hb + lq) * HD_ + c0;
    const float* krow = kh + (hb + lk) * HD_ + c0;
    const float* vrow = vh + (hb + lk) * HD_ + c0;
    for (int j = 0; j < 32; j += 4) {
      float4 fq = *(const float4*)(qrow + j);
      float4 fk = *(const float4*)(krow + j);
      float4 fv = *(const float4*)(vrow + j);
      _Float16* qd = qs  + r * 72 + c0 + j;
      _Float16* kd = ksS + r * 72 + c0 + j;
      qd[0] = (_Float16)fq.x; qd[1] = (_Float16)fq.y;
      qd[2] = (_Float16)fq.z; qd[3] = (_Float16)fq.w;
      kd[0] = (_Float16)fk.x; kd[1] = (_Float16)fk.y;
      kd[2] = (_Float16)fk.z; kd[3] = (_Float16)fk.w;
      vsT[(c0 + j + 0) * 136 + r] = (_Float16)fv.x;
      vsT[(c0 + j + 1) * 136 + r] = (_Float16)fv.y;
      vsT[(c0 + j + 2) * 136 + r] = (_Float16)fv.z;
      vsT[(c0 + j + 3) * 136 + r] = (_Float16)fv.w;
    }
    if ((tid & 1) == 0) kmf[r] = mask[(size_t)b * L_ + lk] ? 1.0f : 0.0f;
  }
  __syncthreads();

  const int w    = tid >> 5;
  const int lane = tid & 31;
  const int half = lane >> 4;
  const int cl   = lane & 15;

  // scores: each wave owns full rows 16w..16w+15 x 128 cols (8 C tiles)
  v8f zero = {};
  v8f sacc[8];
  for (int t = 0; t < 8; t++) sacc[t] = zero;
  for (int koff = 0; koff < HD_; koff += 32) {
    v16h a = ld_frag_a(qs, 72, 16 * w, koff);
    for (int t = 0; t < 8; t++) {
      v16h bf = ld_frag_b(ksS, 72, 16 * t, koff);
      sacc[t] = __builtin_amdgcn_wmma_f32_16x16x32_f16(
          false, a, false, bf, (short)0, sacc[t], false, false);
    }
  }
  float km[8];
  for (int t = 0; t < 8; t++) km[t] = kmf[16 * t + cl];
  __syncthreads();  // qs/ks dead from here; attn may alias them

  // entmax-1.5 bisection entirely in registers. For fixed i, lanes 0-15 hold
  // row 16w+i and lanes 16-31 hold row 16w+i+8; shfl_xor over d<16 reduces
  // within each 16-lane row group (8 cols/lane * 16 lanes = 128 cols).
  const float scale = 0.125f;                         // 1/sqrt(HD)
  const float dm0   = 1.0f - 0.08838834764831845f;    // 1 - sqrt(1/128)
  for (int i = 0; i < 8; i++) {
    const int r = 16 * w + i + 8 * half;
    float X[8];
    float mx = -3.0e38f;
    for (int t = 0; t < 8; t++) {
      float v = sacc[t][i] * scale;
      if (km[t] == 0.0f) v = -BIGF;
      X[t] = 0.5f * v;
      mx = fmaxf(mx, X[t]);
    }
    for (int d = 1; d < 16; d <<= 1) mx = fmaxf(mx, __shfl_xor(mx, d, 32));
    float tau_lo = mx - 1.0f;
    float dm = dm0;
    for (int it = 0; it < 30; it++) {
      dm *= 0.5f;
      const float tau = tau_lo + dm;
      float s = 0.0f;
      for (int t = 0; t < 8; t++) {
        const float u = fmaxf(X[t] - tau, 0.0f);
        s += u * u;
      }
      for (int d = 1; d < 16; d <<= 1) s += __shfl_xor(s, d, 32);
      if (s - 1.0f >= 0.0f) tau_lo = tau;
    }
    const float tau = tau_lo + 0.5f * dm;
    float pv[8];
    float s = 0.0f;
    for (int t = 0; t < 8; t++) {
      const float u = fmaxf(X[t] - tau, 0.0f);
      pv[t] = u * u;
      s += pv[t];
    }
    for (int d = 1; d < 16; d <<= 1) s += __shfl_xor(s, d, 32);
    const float inv = (s > 0.0f) ? 1.0f / s : 0.0f;
    for (int t = 0; t < 8; t++)
      attn[r * 136 + 16 * t + cl] = (_Float16)(pv[t] * inv);
  }
  __syncthreads();

  // ctx = attn(128x128) @ V(128x64); B staged as vsT[n=hd][k=key]
  v8f cacc[4];
  for (int t = 0; t < 4; t++) cacc[t] = zero;
  for (int koff = 0; koff < BS_; koff += 32) {
    v16h a = ld_frag_a(attn, 136, 16 * w, koff);
    for (int t = 0; t < 4; t++) {
      v16h bf = ld_frag_b(vsT, 136, 16 * t, koff);
      cacc[t] = __builtin_amdgcn_wmma_f32_16x16x32_f16(
          false, a, false, bf, (short)0, cacc[t], false, false);
    }
  }
  // scatter-add with inverse permutation (ctx[idxq[r]] += row r) and mean /P
  for (int i = 0; i < 8; i++) {
    const int r  = 16 * w + i + 8 * half;
    const int lq = idxq[sbase + r];
    float* crow = ctx + ((((size_t)b * H_ + h) * L_) + lq) * HD_;
    for (int t = 0; t < 4; t++)
      atomicAdd(crow + 16 * t + cl, cacc[t][i] * 0.25f);
  }
}

// ---------------------------------------------------------------------------
// 5) finalize: (B,H,L,HD) -> (B,L,D), masked zeroing
// ---------------------------------------------------------------------------
__global__ __launch_bounds__(256) void k_finalize(const float* __restrict__ ctx,
                                                  const uint8_t* __restrict__ mask,
                                                  float* __restrict__ out) {
  const size_t t = (size_t)blockIdx.x * 256 + threadIdx.x;
  if (t >= (size_t)B_ * L_ * D_) return;
  const int n  = (int)(t & (D_ - 1));
  const size_t bl = t >> 10;           // / D_
  const int l  = (int)(bl & (L_ - 1));
  const int b  = (int)(bl >> 12);      // / L_
  const int h  = n >> 6;
  const int hd = n & 63;
  const float v = ctx[(((size_t)b * H_ + h) * L_ + l) * HD_ + hd];
  out[t] = mask[(size_t)b * L_ + l] ? v : 0.0f;
}

// ---------------------------------------------------------------------------
// host launcher
// ---------------------------------------------------------------------------
extern "C" void kernel_launch(void* const* d_in, const int* in_sizes, int n_in,
                              void* d_out, int out_size, void* d_ws, size_t ws_size,
                              hipStream_t stream) {
  (void)in_sizes; (void)n_in; (void)out_size;
  // setup_inputs order: k, v, q, mask, Wq, bq, Wk, bk, Wv, bv, pw_q, pb_q, pw_k, pb_k
  const float*   in_k = (const float*)d_in[0];
  const float*   in_v = (const float*)d_in[1];
  const float*   in_q = (const float*)d_in[2];
  const uint8_t* mask = (const uint8_t*)d_in[3];  // jax bool -> 1 byte/elem
  const float*   Wq   = (const float*)d_in[4];
  const float*   bq   = (const float*)d_in[5];
  const float*   Wk   = (const float*)d_in[6];
  const float*   bk   = (const float*)d_in[7];
  const float*   Wv   = (const float*)d_in[8];
  const float*   bv   = (const float*)d_in[9];
  const float*   pwq  = (const float*)d_in[10];
  const float*   pbq  = (const float*)d_in[11];
  const float*   pwk  = (const float*)d_in[12];
  const float*   pbk  = (const float*)d_in[13];

  const size_t SZ_PROJ = (size_t)B_ * H_ * L_ * HD_ * sizeof(float);  // 32 MB
  const size_t SZ_KEY  = (size_t)B_ * H_ * P_ * L_ * sizeof(float);   //  2 MB
  const size_t OFF_QH   = 0;
  const size_t OFF_KH   = OFF_QH + SZ_PROJ;
  const size_t OFF_VH   = OFF_KH + SZ_PROJ;
  const size_t OFF_CTX  = OFF_VH + SZ_PROJ;
  const size_t OFF_QKEY = OFF_CTX + SZ_PROJ;
  const size_t OFF_KKEY = OFF_QKEY + SZ_KEY;   // contiguous with QKEY
  const size_t OFF_IDXQ = OFF_KKEY + SZ_KEY;
  const size_t OFF_IDXK = OFF_IDXQ + SZ_KEY;   // contiguous with IDXQ
  const size_t WS_NEED  = OFF_IDXK + SZ_KEY;   // ~136 MB
  if (ws_size < WS_NEED) return;

  char* ws = (char*)d_ws;
  float* qh   = (float*)(ws + OFF_QH);
  float* kh   = (float*)(ws + OFF_KH);
  float* vh   = (float*)(ws + OFF_VH);
  float* ctx  = (float*)(ws + OFF_CTX);
  float* qkey = (float*)(ws + OFF_QKEY);
  float* kkey = (float*)(ws + OFF_KKEY);
  int*   idxq = (int*)(ws + OFF_IDXQ);
  int*   idxk = (int*)(ws + OFF_IDXK);

  hipMemsetAsync(ctx, 0, SZ_PROJ, stream);

  dim3 gGemm(64, 8);  // (B*L)/128 x D/128
  k_gemm_proj<<<gGemm, 256, 0, stream>>>(in_q, Wq, bq, qh);
  k_gemm_proj<<<gGemm, 256, 0, stream>>>(in_k, Wk, bk, kh);
  k_gemm_proj<<<gGemm, 256, 0, stream>>>(in_v, Wv, bv, vh);

  k_lowrank<<<(B_ * H_ * L_) / 256, 256, 0, stream>>>(qh, kh, pwq, pbq, pwk, pbk,
                                                      mask, qkey, kkey);

  // 128 q-sorts then 128 k-sorts (qkey/kkey and idxq/idxk are contiguous)
  k_bitonic_argsort<<<2 * B_ * H_ * P_, 1024, 0, stream>>>(qkey, idxq);

  k_block_attn<<<B_ * H_ * P_ * NB_, 256, 0, stream>>>(qh, kh, vh, idxq, idxk,
                                                       mask, ctx);

  k_finalize<<<(B_ * L_ * D_) / 256, 256, 0, stream>>>(ctx, mask, (float*)d_out);
}
